// MultiHeadAttention_16260746182758
// MI455X (gfx1250) — matrix operand order repaired and checked by reading, not compiled, not run on hardware
//
#include <hip/hip_runtime.h>
#include <hip/hip_bf16.h>

// ---------------------------------------------------------------------------
// MHA forward for MI455X (gfx1250, wave32, WMMA 16x16x32 f16 -> f32 acc).
// Compute-bound (~137 GFLOP vs ~150MB fp32 traffic at 23.3 TB/s) -> push all
// matmuls through the WMMA pipe; double-buffer LDS to overlap staging.
// ---------------------------------------------------------------------------

typedef __attribute__((ext_vector_type(16))) _Float16 v16h;
typedef __attribute__((ext_vector_type(8)))  _Float16 v8h;
typedef __attribute__((ext_vector_type(4)))  _Float16 v4h;
typedef __attribute__((ext_vector_type(8)))  float    v8f;

#define D_MODEL 1024
#define N_HEADS 16
#define HEAD_DIM 64
#define BATCH 4
#define SEQ 2048
#define LDT 40   // LDS tile stride (f16): multiple of 8 (16B chunks aligned), conflict-spread
#define BM 128
#define BN 128
#define BK 32

// A-fragment (16x32 f16) per CDNA5 ISA 7.12.2, from row-major [16 x >=32] (ld in f16):
//   lane<16 : row=lane,    K = {0..7, 16..23}
//   lane>=16: row=lane-16, K = {8..15, 24..31}
__device__ __forceinline__ v16h load_fragA(const _Float16* __restrict__ p, int ld) {
    const int lane  = threadIdx.x & 31;
    const int row   = lane & 15;
    const int khalf = lane >> 4;
    const _Float16* base = p + row * ld + khalf * 8;
    v8h lo = *(const v8h*)(base);        // K = khalf*8 .. +7
    v8h hi = *(const v8h*)(base + 16);   // K = 16 + khalf*8 .. +7
    v16h r;
#pragma unroll
    for (int i = 0; i < 8; ++i) { r[i] = lo[i]; r[i + 8] = hi[i]; }
    return r;
}

// B-fragment (32x16 KxN f16) from a TRANSPOSED store Bt[n][k] (row-major, ld in f16):
//   lane<16 : N=lane,    K = {0..7, 16..23} ; lane>=16: N=lane-16, K = {8..15, 24..31}
__device__ __forceinline__ v16h load_fragBt(const _Float16* __restrict__ p, int ld) {
    const int lane = threadIdx.x & 31;
    const int col  = lane & 15;
    const int kb   = (lane >> 4) << 3;
    const _Float16* base = p + col * ld + kb;
    v8h lo = *(const v8h*)(base);
    v8h hi = *(const v8h*)(base + 16);
    v16h r;
#pragma unroll
    for (int i = 0; i < 8; ++i) { r[i] = lo[i]; r[i + 8] = hi[i]; }
    return r;
}

__device__ __forceinline__ v8f wmma_f16(v16h a, v16h b, v8f c) {
    return __builtin_amdgcn_wmma_f32_16x16x32_f16(false, a, false, b, (short)0, c, false, false);
}

// ---------------------------------------------------------------------------
// GEMM: out = A(MxK fp32) @ W^T + bias,  W is [1024,K] fp32 row-major.
//  outF != nullptr : fp32 row-major [M,1024]                      (output proj)
//  outF == nullptr : f16 head-split [B, H, S, HEAD_DIM] * scale   (QKV proj)
// 256 thr / 8 waves; block tile 128x128; wave tile 32x64 (2x4 frags);
// double-buffered LDS, 8 WMMA per wave per K-step of 32.
// ---------------------------------------------------------------------------
__global__ __launch_bounds__(256)
void gemm_wmma_kernel(const float* __restrict__ A, const float* __restrict__ W,
                      const float* __restrict__ bias,
                      float* __restrict__ outF, _Float16* __restrict__ outH,
                      int K, float scale) {
    __shared__ __align__(16) _Float16 As[2][BM * LDT];
    __shared__ __align__(16) _Float16 Bs[2][BM * LDT];   // staged as Bt[n][k]

    const int tid  = threadIdx.x;
    const int wave = tid >> 5;
    const int lane = tid & 31;
    const int wr   = wave & 3;        // row block: 32 rows
    const int wc   = wave >> 2;       // col block: 64 cols
    const int m0   = blockIdx.y * BM;
    const int n0   = blockIdx.x * BN;

    // fp32 -> f16 tile staging: 1024 float4 per matrix, 4 per thread, all b128/b64.
    auto stage = [&](int buf, int k0) {
#pragma unroll
        for (int i = 0; i < 4; ++i) {
            const int f = tid + i * 256;           // float4 id
            const int r = f >> 3, c = (f & 7) << 2;
            const float4 av = *(const float4*)(A + (size_t)(m0 + r) * K + k0 + c);
            const float4 wv = *(const float4*)(W + (size_t)(n0 + r) * K + k0 + c);
            v4h ah, wh;
            ah[0] = (_Float16)av.x; ah[1] = (_Float16)av.y;
            ah[2] = (_Float16)av.z; ah[3] = (_Float16)av.w;
            wh[0] = (_Float16)wv.x; wh[1] = (_Float16)wv.y;
            wh[2] = (_Float16)wv.z; wh[3] = (_Float16)wv.w;
            *(v4h*)&As[buf][r * LDT + c] = ah;
            *(v4h*)&Bs[buf][r * LDT + c] = wh;
        }
    };

    v8f acc[2][4] = {};
    stage(0, 0);
    __syncthreads();

    const int nsteps = K / BK;
    for (int kt = 0; kt < nsteps; ++kt) {
        const int cur = kt & 1;
        if (kt + 1 < nsteps) stage(cur ^ 1, (kt + 1) * BK);
        if (kt + 2 < nsteps) {  // gfx1250 global_prefetch_b8 two tiles ahead
            __builtin_prefetch(A + (size_t)(m0 + (tid >> 1)) * K + (kt + 2) * BK + ((tid & 1) << 4), 0, 0);
            __builtin_prefetch(W + (size_t)(n0 + (tid >> 1)) * K + (kt + 2) * BK + ((tid & 1) << 4), 0, 0);
        }

        const v16h a0 = load_fragA(&As[cur][(wr * 32 +  0) * LDT], LDT);
        const v16h a1 = load_fragA(&As[cur][(wr * 32 + 16) * LDT], LDT);
        v16h b[4];
#pragma unroll
        for (int nt = 0; nt < 4; ++nt)
            b[nt] = load_fragBt(&Bs[cur][(wc * 64 + nt * 16) * LDT], LDT);
#pragma unroll
        for (int nt = 0; nt < 4; ++nt) {
            acc[0][nt] = wmma_f16(a0, b[nt], acc[0][nt]);
            acc[1][nt] = wmma_f16(a1, b[nt], acc[1][nt]);
        }
        __syncthreads();
    }

    // Epilogue.  C layout: VGPR r -> M = r + 8*(lane>=16), N = lane&15.
    const int g = lane >> 4, n = lane & 15;
#pragma unroll
    for (int mi = 0; mi < 2; ++mi) {
#pragma unroll
        for (int nt = 0; nt < 4; ++nt) {
            const int ncol = n0 + wc * 64 + nt * 16 + n;
            const float bv = bias[ncol];
#pragma unroll
            for (int r = 0; r < 8; ++r) {
                const int m = m0 + wr * 32 + mi * 16 + r + 8 * g;
                const float v = (acc[mi][nt][r] + bv) * scale;
                if (outF) {
                    outF[(size_t)m * D_MODEL + ncol] = v;
                } else {
                    const int b = m >> 11, s = m & (SEQ - 1);
                    const int h = ncol >> 6, d = ncol & (HEAD_DIM - 1);
                    outH[(((size_t)(b * N_HEADS + h) * SEQ) + s) * HEAD_DIM + d] = (_Float16)v;
                }
            }
        }
    }
}

// ---------------------------------------------------------------------------
// Flash attention: grid = (S/64 q-tiles, B*H). 128 thr (4 waves), wave = 16 q
// rows. Q pre-scaled by 1/sqrt(64). 16 WMMA per wave per 64-key tile.
// ---------------------------------------------------------------------------
__global__ __launch_bounds__(128)
void attention_kernel(const _Float16* __restrict__ Qh, const _Float16* __restrict__ Kh,
                      const _Float16* __restrict__ Vh, float* __restrict__ ctx) {
    __shared__ __align__(16) _Float16 Vs[64 * 72];       // V transposed: Vs[d][key]
    __shared__ __align__(16) _Float16 Ps[4][16 * 72];    // per-wave P tile [q][key]

    const int tid = threadIdx.x, wave = tid >> 5, lane = tid & 31;
    const int bh  = blockIdx.y;                 // b*16 + h
    const int q0  = blockIdx.x * 64 + wave * 16;
    const size_t baseQ  = ((size_t)bh * SEQ + q0) * HEAD_DIM;
    const size_t baseKV = (size_t)bh * SEQ * HEAD_DIM;

    v16h qf[2];
    qf[0] = load_fragA(Qh + baseQ,      HEAD_DIM);   // d 0..31
    qf[1] = load_fragA(Qh + baseQ + 32, HEAD_DIM);   // d 32..63

    v8f acc[4] = {};
    float m_row[8], l_row[8];
#pragma unroll
    for (int r = 0; r < 8; ++r) { m_row[r] = -1e30f; l_row[r] = 0.f; }

    const int g = lane >> 4, n = lane & 15;
    _Float16* myP = &Ps[wave][0];

    for (int kt = 0; kt < SEQ / 64; ++kt) {
        const int kv0 = kt * 64;
        __syncthreads();  // previous iteration's Vs reads complete
        // stage V transposed: 512 v8h loads, 4 per thread, fixed trip count
#pragma unroll
        for (int i = 0; i < 4; ++i) {
            const int it  = tid + i * 128;
            const int key = it >> 3, dc = (it & 7) << 3;
            const v8h chunk = *(const v8h*)(Vh + baseKV + (size_t)(kv0 + key) * HEAD_DIM + dc);
#pragma unroll
            for (int e = 0; e < 8; ++e) Vs[(dc + e) * 72 + key] = chunk[e];
        }

        // scores: S = Q @ K^T  (K row-major [key,d] is exactly Bt layout for K^T)
        v8f sf[4] = {};
        v16h kf[4][2];
#pragma unroll
        for (int nt = 0; nt < 4; ++nt) {
            const _Float16* kp = Kh + baseKV + (size_t)(kv0 + nt * 16) * HEAD_DIM;
            kf[nt][0] = load_fragBt(kp,      HEAD_DIM);
            kf[nt][1] = load_fragBt(kp + 32, HEAD_DIM);
        }
#pragma unroll
        for (int nt = 0; nt < 4; ++nt) {
            sf[nt] = wmma_f16(qf[0], kf[nt][0], sf[nt]);
            sf[nt] = wmma_f16(qf[1], kf[nt][1], sf[nt]);
        }

        // online softmax; rows r / r+8 per lane-half, 16-lane reductions stay in-half
#pragma unroll
        for (int r = 0; r < 8; ++r) {
            float v = fmaxf(fmaxf(sf[0][r], sf[1][r]), fmaxf(sf[2][r], sf[3][r]));
#pragma unroll
            for (int off = 1; off < 16; off <<= 1) v = fmaxf(v, __shfl_xor(v, off, 32));
            const float mnew = fmaxf(m_row[r], v);
            const float corr = __expf(m_row[r] - mnew);
            m_row[r] = mnew;
            l_row[r] *= corr;
#pragma unroll
            for (int j = 0; j < 4; ++j) acc[j][r] *= corr;
            float rs = 0.f;
#pragma unroll
            for (int nt = 0; nt < 4; ++nt) {
                const float p = __expf(sf[nt][r] - mnew);
                rs += p;
                myP[(r + 8 * g) * 72 + nt * 16 + n] = (_Float16)p;
            }
#pragma unroll
            for (int off = 1; off < 16; off <<= 1) rs += __shfl_xor(rs, off, 32);
            l_row[r] += rs;
        }
        __syncthreads();  // Vs staged + P stores visible

        // O += P @ V   (A = P tile from LDS, B = V^T tile from LDS)
#pragma unroll
        for (int ks = 0; ks < 2; ++ks) {
            const v16h pa = load_fragA(myP + ks * 32, 72);
#pragma unroll
            for (int j = 0; j < 4; ++j) {
                const v16h vf = load_fragBt(Vs + (j * 16) * 72 + ks * 32, 72);
                acc[j] = wmma_f16(pa, vf, acc[j]);
            }
        }
    }

    // normalize and write context [b, s, h*64+d] (fp32) for the output GEMM
    const int b = bh >> 4, h = bh & 15;
#pragma unroll
    for (int r = 0; r < 8; ++r) {
        const int q = q0 + r + 8 * g;
        const float inv = 1.0f / l_row[r];
#pragma unroll
        for (int j = 0; j < 4; ++j)
            ctx[((size_t)b * SEQ + q) * D_MODEL + h * HEAD_DIM + j * 16 + n] = acc[j][r] * inv;
    }
}

// ---------------------------------------------------------------------------
extern "C" void kernel_launch(void* const* d_in, const int* in_sizes, int n_in,
                              void* d_out, int out_size, void* d_ws, size_t ws_size,
                              hipStream_t stream) {
    (void)in_sizes; (void)n_in; (void)out_size; (void)ws_size;
    const float* q  = (const float*)d_in[0];
    const float* k  = (const float*)d_in[1];
    const float* v  = (const float*)d_in[2];
    const float* Wq = (const float*)d_in[3];
    const float* bq = (const float*)d_in[4];
    const float* Wk = (const float*)d_in[5];
    const float* bk = (const float*)d_in[6];
    const float* Wv = (const float*)d_in[7];
    const float* bv = (const float*)d_in[8];
    const float* Wo = (const float*)d_in[9];
    const float* bo = (const float*)d_in[10];

    char* ws = (char*)d_ws;
    _Float16* Qh  = (_Float16*)(ws);                            // 16 MiB
    _Float16* Kh  = (_Float16*)(ws + ((size_t)16 << 20));       // 16 MiB
    _Float16* Vh  = (_Float16*)(ws + ((size_t)32 << 20));       // 16 MiB
    float*    ctx = (float*)   (ws + ((size_t)48 << 20));       // 32 MiB

    const dim3 blk256(256), blk128(128);
    const dim3 gproj(D_MODEL / BN, (BATCH * SEQ) / BM);         // 8 x 64
    const dim3 gattn(SEQ / 64, BATCH * N_HEADS);                // 32 x 64

    gemm_wmma_kernel<<<gproj, blk256, 0, stream>>>(q, Wq, bq, nullptr, Qh, D_MODEL, 0.125f);
    gemm_wmma_kernel<<<gproj, blk256, 0, stream>>>(k, Wk, bk, nullptr, Kh, D_MODEL, 1.0f);
    gemm_wmma_kernel<<<gproj, blk256, 0, stream>>>(v, Wv, bv, nullptr, Vh, D_MODEL, 1.0f);
    attention_kernel<<<gattn, blk128, 0, stream>>>(Qh, Kh, Vh, ctx);
    gemm_wmma_kernel<<<gproj, blk256, 0, stream>>>(ctx, Wo, bo, (float*)d_out, nullptr, D_MODEL, 1.0f);
}
